// RCMambaBlock_32315333935392
// MI455X (gfx1250) — matrix operand order repaired and checked by reading, not compiled, not run on hardware
//
#include <hip/hip_runtime.h>
#include <hip/hip_bf16.h>

#define D_MODEL 1024
#define D_STATE 16
#define D_INNER 2048
#define DT_RANK 64
#define D_CONV 4
#define R_DIM 512
#define FILM_HID 32
#define B_SZ 2
#define SEQ_L 2048
#define DBC_N (DT_RANK + 2 * D_STATE)   // 96 (logical)
#define DBC_P 128                        // padded row stride for dbc / W_x_pad

typedef __bf16 bf16_t;
typedef __attribute__((ext_vector_type(16))) __bf16 v16bf;
typedef __attribute__((ext_vector_type(4)))  __bf16 v4bf;
typedef __attribute__((ext_vector_type(8)))  float  v8f;

union Frag {
    v16bf v;
    unsigned int u[8];
};

// ---------------------------------------------------------------------------
// bf16-WMMA GEMM: C(MxN) = A(MxK, stride lda) * B(KxN, stride ldb).
// REQUIRES: M%128==0, N%64==0, K%32==0, lda/ldb/ldc%4==0 -> branchless.
// Block = 256 threads = 8 waves; block tile 128x64; each wave computes a
// 32x32 register tile = 4 x v_wmma_f32_16x16x32_bf16 per K-step.
// Double-buffered LDS, ONE barrier per K-step, t+1 loads in regs, t+2
// prefetched (global_prefetch_b8).
// ---------------------------------------------------------------------------
#define BM 128
#define BN 64
#define BKK 32
#define LDA_S 36    // padded bf16 row stride (even -> aligned pair loads)
#define LDBT_S 36

__global__ __launch_bounds__(256) void wmma_gemm_bf16_kernel(
    const float* __restrict__ A, const float* __restrict__ B,
    float* __restrict__ C, int K, int lda, int ldb, int ldc)
{
    __shared__ __align__(16) bf16_t lA[2][BM * LDA_S];    // [row][k]
    __shared__ __align__(16) bf16_t lBt[2][BN * LDBT_S];  // [col][k] (B^T)

    const int tid  = threadIdx.x;
    const int lane = tid & 31;
    const int w    = tid >> 5;        // wave 0..7
    const int wm   = w >> 1;          // 0..3 -> 32-row strip
    const int wn   = w & 1;           // 0..1 -> 32-col strip
    const int blockM = blockIdx.y * BM;
    const int blockN = blockIdx.x * BN;

    const int r  = lane & 15;         // row (A) / col (B) within 16
    const int hi = lane >> 4;         // half-wave select per ISA layout

    // staging coordinates: A = 4 float4 / thread, B = 2 float4 / thread
    const int arow = tid >> 3;                // 0..31 (strips of 32)
    const int acol = (tid & 7) * 4;           // 0..28
    const int br0  = tid >> 4;                // k row 0..15
    const int bc0  = (tid & 15) * 4;          // n col 0..60
    const int br1  = br0 + 16;                // k row 16..31

    const float* aP = &A[(long)(blockM + arow) * lda + acol];
    const float* bP0 = &B[(long)br0 * ldb + blockN + bc0];
    const float* bP1 = &B[(long)br1 * ldb + blockN + bc0];
    const long aRowStride = (long)32 * lda;   // strip step

    v8f acc00 = {}, acc01 = {}, acc10 = {}, acc11 = {};

    const int nt = K / BKK;
    float4 fA[4], fB0, fB1;

    // --- preload tile 0 ---
#pragma unroll
    for (int j = 0; j < 4; ++j) fA[j] = *(const float4*)(aP + j * aRowStride);
    fB0 = *(const float4*)(bP0);
    fB1 = *(const float4*)(bP1);

    for (int t = 0; t < nt; ++t) {
        const int cur = t & 1;
        // --- store staged registers to LDS buffer `cur` ---
#pragma unroll
        for (int j = 0; j < 4; ++j) {
            v4bf pa = { (bf16_t)fA[j].x, (bf16_t)fA[j].y,
                        (bf16_t)fA[j].z, (bf16_t)fA[j].w };
            *(v4bf*)&lA[cur][(arow + j * 32) * LDA_S + acol] = pa;
        }
        // B transposed scatter: lBt[n][k]
        lBt[cur][(bc0 + 0) * LDBT_S + br0] = (bf16_t)fB0.x;
        lBt[cur][(bc0 + 1) * LDBT_S + br0] = (bf16_t)fB0.y;
        lBt[cur][(bc0 + 2) * LDBT_S + br0] = (bf16_t)fB0.z;
        lBt[cur][(bc0 + 3) * LDBT_S + br0] = (bf16_t)fB0.w;
        lBt[cur][(bc0 + 0) * LDBT_S + br1] = (bf16_t)fB1.x;
        lBt[cur][(bc0 + 1) * LDBT_S + br1] = (bf16_t)fB1.y;
        lBt[cur][(bc0 + 2) * LDBT_S + br1] = (bf16_t)fB1.z;
        lBt[cur][(bc0 + 3) * LDBT_S + br1] = (bf16_t)fB1.w;

        __syncthreads();   // single barrier per K-step (double-buffered)

        // --- issue global loads for tile t+1 (overlap with WMMA below) ---
        if (t + 1 < nt) {
            const long ko = (long)(t + 1) * BKK;
#pragma unroll
            for (int j = 0; j < 4; ++j)
                fA[j] = *(const float4*)(aP + ko + j * aRowStride);
            fB0 = *(const float4*)(bP0 + ko * ldb);
            fB1 = *(const float4*)(bP1 + ko * ldb);
        }
        // --- prefetch tile t+2 toward caches (global_prefetch_b8) ---
        if (t + 2 < nt) {
            const long kp = (long)(t + 2) * BKK;
            __builtin_prefetch(aP + kp, 0, 1);
            __builtin_prefetch(bP0 + kp * ldb, 0, 1);
        }

        // --- A fragments (16x32 bf16, ISA 16-bit A layout) ---
        Frag fa0, fa1;
#pragma unroll
        for (int v = 0; v < 8; ++v) {
            int kb = 2 * v + 8 * hi + ((v >= 4) ? 8 : 0);  // even K base
            fa0.u[v] = *(const unsigned int*)
                &lA[cur][(wm * 32 + r) * LDA_S + kb];
            fa1.u[v] = *(const unsigned int*)
                &lA[cur][(wm * 32 + 16 + r) * LDA_S + kb];
        }
        // --- B fragments (32x16 bf16, ISA 16-bit B layout) ---
        Frag fb0, fb1;
#pragma unroll
        for (int v = 0; v < 8; ++v) {
            int kk = 2 * v + 8 * hi + ((v >= 4) ? 8 : 0);  // even K base
            fb0.u[v] = *(const unsigned int*)
                &lBt[cur][(wn * 32 + r) * LDBT_S + kk];
            fb1.u[v] = *(const unsigned int*)
                &lBt[cur][(wn * 32 + 16 + r) * LDBT_S + kk];
        }

        acc00 = __builtin_amdgcn_wmma_f32_16x16x32_bf16(
                    false, fa0.v, false, fb0.v, (short)0, acc00, false, false);
        acc01 = __builtin_amdgcn_wmma_f32_16x16x32_bf16(
                    false, fa0.v, false, fb1.v, (short)0, acc01, false, false);
        acc10 = __builtin_amdgcn_wmma_f32_16x16x32_bf16(
                    false, fa1.v, false, fb0.v, (short)0, acc10, false, false);
        acc11 = __builtin_amdgcn_wmma_f32_16x16x32_bf16(
                    false, fa1.v, false, fb1.v, (short)0, acc11, false, false);
        // no trailing barrier needed: next iter stores hit the other buffer,
        // and the next barrier orders reuse two iterations out.
    }

    // --- epilogue: C/D layout col = lane%16, row = 8*hi + v ---
    const int mB0 = blockM + wm * 32 + 8 * hi;
    const int mB1 = mB0 + 16;
    const int n0 = blockN + wn * 32 + r;
    const int n1 = n0 + 16;
#pragma unroll
    for (int v = 0; v < 8; ++v) {
        C[(long)(mB0 + v) * ldc + n0] = acc00[v];
        C[(long)(mB0 + v) * ldc + n1] = acc01[v];
        C[(long)(mB1 + v) * ldc + n0] = acc10[v];
        C[(long)(mB1 + v) * ldc + n1] = acc11[v];
    }
}

// ---------------------------------------------------------------------------
// Zero-pad W_x (D_INNER x 96) -> W_x_pad (D_INNER x 128)
// ---------------------------------------------------------------------------
__global__ void pad_wx_kernel(const float* __restrict__ W_x,
                              float* __restrict__ W_x_pad)
{
    int i = blockIdx.x * 256 + threadIdx.x;       // over D_INNER*DBC_P
    if (i >= D_INNER * DBC_P) return;
    int k = i >> 7;           // / 128
    int n = i & (DBC_P - 1);
    W_x_pad[i] = (n < DBC_N) ? W_x[k * DBC_N + n] : 0.0f;
}

// ---------------------------------------------------------------------------
// FiLM: film[b, 0:64] = silu(retrieval @ W_f1 + b_f1) @ W_f2 + b_f2
// ---------------------------------------------------------------------------
__global__ void film_kernel(const float* __restrict__ retrieval,
                            const float* __restrict__ W_f1,
                            const float* __restrict__ b_f1,
                            const float* __restrict__ W_f2,
                            const float* __restrict__ b_f2,
                            float* __restrict__ film)
{
    int b = blockIdx.x;
    int t = threadIdx.x;  // 0..31
    __shared__ float hid[FILM_HID];
    float acc = b_f1[t];
    for (int k = 0; k < R_DIM; ++k)
        acc = fmaf(retrieval[b * R_DIM + k], W_f1[k * FILM_HID + t], acc);
    hid[t] = acc / (1.0f + __expf(-acc));   // silu
    __syncthreads();
    for (int j = t; j < 4 * D_STATE; j += 32) {
        float o = b_f2[j];
        for (int k = 0; k < FILM_HID; ++k)
            o = fmaf(hid[k], W_f2[k * 4 * D_STATE + j], o);
        film[b * 4 * D_STATE + j] = o;
    }
}

// ---------------------------------------------------------------------------
// Depthwise causal conv (k=4) + bias + silu, 4 channels per thread (float4).
// x_in = xz[..., :D_INNER].
// ---------------------------------------------------------------------------
__global__ void conv_silu_kernel(const float* __restrict__ xz,
                                 const float* __restrict__ conv_w,
                                 const float* __restrict__ conv_b,
                                 float* __restrict__ u)
{
    const int DQ = D_INNER / 4;
    long i = (long)blockIdx.x * blockDim.x + threadIdx.x;
    if (i >= (long)B_SZ * SEQ_L * DQ) return;
    int d4 = (int)(i % DQ) * 4;
    long row = i / DQ;                   // b*L + l
    int l = (int)(row % SEQ_L);
    long bbase = row - l;                // b*L
    float4 acc = *(const float4*)&conv_b[d4];
    float4 w0 = *(const float4*)&conv_w[(d4 + 0) * D_CONV];  // taps for ch d4
    float4 w1 = *(const float4*)&conv_w[(d4 + 1) * D_CONV];
    float4 w2 = *(const float4*)&conv_w[(d4 + 2) * D_CONV];
    float4 w3 = *(const float4*)&conv_w[(d4 + 3) * D_CONV];
    const float* tap0 = &w0.x; const float* tap1 = &w1.x;
    const float* tap2 = &w2.x; const float* tap3 = &w3.x;
#pragma unroll
    for (int k = 0; k < D_CONV; ++k) {
        int ll = l - (D_CONV - 1) + k;
        if (ll >= 0) {
            float4 xv = *(const float4*)&xz[(bbase + ll) * (2 * D_INNER) + d4];
            acc.x = fmaf(tap0[k], xv.x, acc.x);
            acc.y = fmaf(tap1[k], xv.y, acc.y);
            acc.z = fmaf(tap2[k], xv.z, acc.z);
            acc.w = fmaf(tap3[k], xv.w, acc.w);
        }
    }
    float4 o;
    o.x = acc.x / (1.0f + __expf(-acc.x));
    o.y = acc.y / (1.0f + __expf(-acc.y));
    o.z = acc.z / (1.0f + __expf(-acc.z));
    o.w = acc.w / (1.0f + __expf(-acc.w));
    *(float4*)&u[row * D_INNER + d4] = o;
}

// ---------------------------------------------------------------------------
// delta = softplus(delta_pre + b_dt[d])   (in place, float4)
// ---------------------------------------------------------------------------
__device__ __forceinline__ float softplus1(float v) {
    return fmaxf(v, 0.0f) + log1pf(__expf(-fabsf(v)));
}
__global__ void softplus_bias_kernel(float* __restrict__ delta,
                                     const float* __restrict__ b_dt)
{
    const int DQ = D_INNER / 4;
    long i = (long)blockIdx.x * blockDim.x + threadIdx.x;
    if (i >= (long)B_SZ * SEQ_L * DQ) return;
    int d4 = (int)(i % DQ) * 4;
    float4 v = *(float4*)&delta[i * 4];
    float4 bb = *(const float4*)&b_dt[d4];
    float4 o;
    o.x = softplus1(v.x + bb.x);
    o.y = softplus1(v.y + bb.y);
    o.z = softplus1(v.z + bb.z);
    o.w = softplus1(v.w + bb.w);
    *(float4*)&delta[i * 4] = o;
}

// ---------------------------------------------------------------------------
// Fused SSM scan: one (b,d) channel per 16-lane half-group; lane = state n.
// h = exp(delta*A)*h + delta*B_mod*u ; y = <h, C_mod> (shfl_xor reduce over
// the 16 states) ; output ygate = (y + u*D) * silu(z).
// ---------------------------------------------------------------------------
__global__ __launch_bounds__(256) void ssm_scan_kernel(
    const float* __restrict__ dbc,    // (B*L, 128): [64:80]=B_ssm, [80:96]=C_ssm
    const float* __restrict__ delta,  // (B*L, D_INNER) after softplus
    const float* __restrict__ u,      // (B*L, D_INNER)
    const float* __restrict__ xz,     // (B*L, 2*D_INNER); z = [..., D_INNER:]
    const float* __restrict__ A_log,  // (D_INNER, D_STATE)
    const float* __restrict__ Dvec,   // (D_INNER)
    const float* __restrict__ film,   // (B, 64) = gB,bB,gC,bC
    float* __restrict__ yg)           // (B*L, D_INNER)
{
    const int grp = threadIdx.x >> 4;           // channel group within block
    const int n   = threadIdx.x & 15;           // state index
    const int gch = blockIdx.x * 16 + grp;      // 0 .. B*D_INNER-1
    const int b   = gch / D_INNER;
    const int d   = gch % D_INNER;

    const float An = -__expf(A_log[d * D_STATE + n]);
    const float Dd = Dvec[d];
    const float* f = film + b * 4 * D_STATE;
    const float gB = f[n];
    const float bB = f[D_STATE + n];
    const float gC = f[2 * D_STATE + n];
    const float bC = f[3 * D_STATE + n];

    float h = 0.0f;
    const long base = (long)b * SEQ_L;
    for (int l = 0; l < SEQ_L; ++l) {
        long row = base + l;
        float dlt = delta[row * D_INNER + d];
        float uv  = u[row * D_INNER + d];
        float Bn  = dbc[row * DBC_P + DT_RANK + n];
        float Cn  = dbc[row * DBC_P + DT_RANK + D_STATE + n];
        float Bm  = fmaf(gB, Bn, Bn) + bB;      // (1+gB)*B + bB
        float Cm  = fmaf(gC, Cn, Cn) + bC;      // (1+gC)*C + bC
        float dA  = __expf(dlt * An);
        h = fmaf(dA, h, dlt * Bm * uv);
        float p = h * Cm;
        // butterfly reduce over the 16 states (stays inside 16-lane group)
        p += __shfl_xor(p, 1, 32);
        p += __shfl_xor(p, 2, 32);
        p += __shfl_xor(p, 4, 32);
        p += __shfl_xor(p, 8, 32);
        if (n == 0) {
            float zv = xz[row * (2 * D_INNER) + D_INNER + d];
            float sz = zv / (1.0f + __expf(-zv));   // silu(z)
            yg[row * D_INNER + d] = (p + uv * Dd) * sz;
        }
    }
}

// ---------------------------------------------------------------------------
extern "C" void kernel_launch(void* const* d_in, const int* in_sizes, int n_in,
                              void* d_out, int out_size, void* d_ws, size_t ws_size,
                              hipStream_t stream)
{
    const float* x         = (const float*)d_in[0];
    const float* retrieval = (const float*)d_in[1];
    const float* W_in      = (const float*)d_in[2];
    const float* conv_w    = (const float*)d_in[3];
    const float* conv_b    = (const float*)d_in[4];
    const float* W_x       = (const float*)d_in[5];
    const float* W_dt      = (const float*)d_in[6];
    const float* b_dt      = (const float*)d_in[7];
    const float* A_log     = (const float*)d_in[8];
    const float* Dvec      = (const float*)d_in[9];
    const float* W_out     = (const float*)d_in[10];
    const float* W_f1      = (const float*)d_in[11];
    const float* b_f1      = (const float*)d_in[12];
    const float* W_f2      = (const float*)d_in[13];
    const float* b_f2      = (const float*)d_in[14];
    float* out = (float*)d_out;

    const long BL = (long)B_SZ * SEQ_L;                 // 4096 rows
    char* ws = (char*)d_ws;
    size_t off = 0;
    float* film   = (float*)(ws + off); off += 1024;
    float* W_x_pad= (float*)(ws + off); off += (size_t)D_INNER * DBC_P * 4;     // 1 MB
    float* xz     = (float*)(ws + off); off += (size_t)BL * 2 * D_INNER * 4;    // 67 MB
    float* u      = (float*)(ws + off); off += (size_t)BL * D_INNER * 4;        // 33.5 MB
    float* dbc    = (float*)(ws + off); off += (size_t)BL * DBC_P * 4;          // 2 MB
    float* delta  = (float*)(ws + off); off += (size_t)BL * D_INNER * 4;        // 33.5 MB
    float* yg     = (float*)(ws + off); off += (size_t)BL * D_INNER * 4;        // 33.5 MB
    (void)ws_size; (void)in_sizes; (void)n_in; (void)out_size;

    // 0) pad W_x -> (D_INNER, 128)
    pad_wx_kernel<<<(D_INNER * DBC_P) / 256, 256, 0, stream>>>(W_x, W_x_pad);

    // 1) FiLM (tiny)
    film_kernel<<<B_SZ, 32, 0, stream>>>(retrieval, W_f1, b_f1, W_f2, b_f2, film);

    // 2) xz = x @ W_in   (4096 x 1024 x 4096)
    wmma_gemm_bf16_kernel<<<dim3((2 * D_INNER) / BN, BL / BM), 256, 0, stream>>>(
        x, W_in, xz, D_MODEL, D_MODEL, 2 * D_INNER, 2 * D_INNER);

    // 3) depthwise conv + silu -> u
    {
        long total = BL * (D_INNER / 4);
        conv_silu_kernel<<<(unsigned)((total + 255) / 256), 256, 0, stream>>>(
            xz, conv_w, conv_b, u);
    }

    // 4) dbc = u @ W_x_pad   (4096 x 2048 x 128)
    wmma_gemm_bf16_kernel<<<dim3(DBC_P / BN, BL / BM), 256, 0, stream>>>(
        u, W_x_pad, dbc, D_INNER, D_INNER, DBC_P, DBC_P);

    // 5) delta_pre = dt_r @ W_dt  (A = dbc[:, :64] via lda = 128)
    wmma_gemm_bf16_kernel<<<dim3(D_INNER / BN, BL / BM), 256, 0, stream>>>(
        dbc, W_dt, delta, DT_RANK, DBC_P, D_INNER, D_INNER);

    // 6) delta = softplus(delta_pre + b_dt)
    {
        long total = BL * (D_INNER / 4);
        softplus_bias_kernel<<<(unsigned)((total + 255) / 256), 256, 0, stream>>>(
            delta, b_dt);
    }

    // 7) fused FiLM-modulated selective scan + gate
    ssm_scan_kernel<<<(B_SZ * D_INNER) / 16, 256, 0, stream>>>(
        dbc, delta, u, xz, A_log, Dvec, film, yg);

    // 8) out = ygate @ W_out  (4096 x 2048 x 1024)
    wmma_gemm_bf16_kernel<<<dim3(D_MODEL / BN, BL / BM), 256, 0, stream>>>(
        yg, W_out, out, D_INNER, D_INNER, D_MODEL, D_MODEL);
}
